// CondMeshGraphNet_32169305047411
// MI455X (gfx1250) — matrix operand order repaired and checked by reading, not compiled, not run on hardware
//
#include <hip/hip_runtime.h>
#include <stdint.h>

// ---------------------------------------------------------------------------
// Types for CDNA5 WMMA (wave32): D(16x16 f32) = A(16x32 bf16) x B(32x16 bf16) + C
// ---------------------------------------------------------------------------
typedef __attribute__((ext_vector_type(16))) __bf16 v16bf;
typedef __attribute__((ext_vector_type(8)))  float  v8f;

union FragA { v16bf v; uint4 u[2]; };

#define WG 256
#define AP 520   // LDS pitch for A tile (bf16 elems): max Dpad 512 + 8 pad (bank skew)
#define HP 136   // LDS pitch for hidden tile: 128 + 8

// ---------------------------------------------------------------------------
// Generic fused 2-layer MLP:  out = [res +] relu(concat(segs) @ W1 + b1) @ W2 + b2
// Segments implement the gather/concat (edge: h[row],h[col],e,u[eb]; node: h,agg,u[batch]).
// ---------------------------------------------------------------------------
struct Seg { const float* src; const int* idx; int cols; int span; };

struct MlpArgs {
  Seg seg[4];
  int nseg;
  int M;                        // number of rows
  int Dpad;                     // padded K1 (multiple of 32, <= 512)
  const unsigned short* W1p;    // packed bf16 B-fragments, (Dpad/32)*8 blocks of 512
  const float* b1;              // [128]
  const unsigned short* W2p;    // packed, 4*ntiles2 blocks of 512
  const float* b2;              // [Dout]
  int Dout;
  int ntiles2;                  // ceil(Dout/16)
  const float* res;             // residual (nullptr if none), stride = ostride
  float* out;
  int ostride;
};

// Pack an f32 row-major weight [K, Ncols] into WMMA bf16 B-fragment blocks.
// Block (kc, nt): lane l holds column nt*16 + (l%16); elem i holds K row
// kc*32 + (l/16)*8 + (i<8 ? i : 16+i-8). Zero-padded outside [K, Ncols].
__global__ __launch_bounds__(WG) void pack_w_kernel(const float* __restrict__ W,
                                                    unsigned short* __restrict__ out,
                                                    int K, int Ncols, int Ntiles,
                                                    int nBlocks) {
  int gw   = blockIdx.x * (WG / 32) + (threadIdx.x >> 5);
  int lane = threadIdx.x & 31;
  if (gw >= nBlocks) return;
  int kc = gw / Ntiles, nt = gw % Ntiles;
  int n = nt * 16 + (lane & 15);
  int kbase = kc * 32 + (lane >> 4) * 8;
  unsigned short* o = out + (size_t)gw * 512 + lane * 16;
#pragma unroll
  for (int i = 0; i < 16; ++i) {
    int k = kbase + (i < 8 ? i : i + 8);
    float v = (k < K && n < Ncols) ? W[(size_t)k * Ncols + n] : 0.0f;
    __bf16 b = (__bf16)v;
    o[i] = __builtin_bit_cast(unsigned short, b);
  }
}

__global__ __launch_bounds__(WG) void mlp2_kernel(MlpArgs A) {
  __shared__ __bf16 sa[32 * AP];
  __shared__ __bf16 sh[32 * HP];
  const int tid   = threadIdx.x;
  const int lane  = tid & 31;
  const int w     = tid >> 5;      // wave 0..7 -> N-tile
  const int lmod  = lane & 15;
  const int lhalf = lane >> 4;
  const int mbase = blockIdx.x * 32;

  // -------- Phase A: gather + concat + f32->bf16 stage into LDS ------------
  for (int c0 = tid; c0 < A.Dpad; c0 += WG) {
    const float* src = nullptr; const int* idx = nullptr; int cols = 0; int cc = c0;
    bool valid = false;
#pragma unroll
    for (int s = 0; s < 4; ++s) {
      if (s < A.nseg) {
        if (cc < A.seg[s].span) {
          if (cc < A.seg[s].cols) {
            src = A.seg[s].src; idx = A.seg[s].idx; cols = A.seg[s].cols; valid = true;
          }
          break;
        }
        cc -= A.seg[s].span;
      }
    }
#pragma unroll 4
    for (int r = 0; r < 32; ++r) {
      int rg = mbase + r;
      float v = 0.0f;
      if (valid && rg < A.M) {
        int rr = idx ? idx[rg] : rg;
        v = src[(size_t)rr * cols + cc];
      }
      sa[r * AP + c0] = (__bf16)v;
    }
  }
  __syncthreads();

  // -------- Phase B: GEMM1 (Dpad x 128), wave w computes cols [16w,16w+16) --
  v8f acc0 = {}; v8f acc1 = {};
  const int nchunks = A.Dpad >> 5;
  for (int kc = 0; kc < nchunks; ++kc) {
    FragA a0, a1, bfr;
    const int koff = (kc << 5) + lhalf * 8;        // A elems 0..7 -> K koff..koff+7
    const __bf16* pa0 = &sa[lmod * AP + koff];
    a0.u[0] = *(const uint4*)(pa0);
    a0.u[1] = *(const uint4*)(pa0 + 16);           // elems 8..15 -> K koff+16..+23
    const __bf16* pa1 = &sa[(lmod + 16) * AP + koff];
    a1.u[0] = *(const uint4*)(pa1);
    a1.u[1] = *(const uint4*)(pa1 + 16);
    const uint4* pb = (const uint4*)(A.W1p + ((size_t)(kc * 8 + w)) * 512 + lane * 16);
    bfr.u[0] = pb[0]; bfr.u[1] = pb[1];
    acc0 = __builtin_amdgcn_wmma_f32_16x16x32_bf16(false, a0.v, false, bfr.v,
                                                   (short)0, acc0, false, false);
    acc1 = __builtin_amdgcn_wmma_f32_16x16x32_bf16(false, a1.v, false, bfr.v,
                                                   (short)0, acc1, false, false);
  }

  // -------- Epilogue 1: bias + ReLU -> hidden (bf16) in LDS ----------------
  {
    const int col = w * 16 + lmod;                 // C layout: N = lane%16
    const float b = A.b1[col];
#pragma unroll
    for (int r = 0; r < 8; ++r) {
      int m = r + lhalf * 8;                       // M = vgpr + 8*(lane>=16)
      float v0 = acc0[r] + b; v0 = v0 > 0.0f ? v0 : 0.0f;
      sh[m * HP + col] = (__bf16)v0;
      float v1 = acc1[r] + b; v1 = v1 > 0.0f ? v1 : 0.0f;
      sh[(m + 16) * HP + col] = (__bf16)v1;
    }
  }
  __syncthreads();

  // -------- Phase C: GEMM2 (128 x Dout) + bias + residual + store ----------
  if (w < A.ntiles2) {
    v8f c0 = {}; v8f c1 = {};
#pragma unroll
    for (int kc = 0; kc < 4; ++kc) {
      FragA a0, a1, bfr;
      const int koff = (kc << 5) + lhalf * 8;
      const __bf16* pa0 = &sh[lmod * HP + koff];
      a0.u[0] = *(const uint4*)(pa0);
      a0.u[1] = *(const uint4*)(pa0 + 16);
      const __bf16* pa1 = &sh[(lmod + 16) * HP + koff];
      a1.u[0] = *(const uint4*)(pa1);
      a1.u[1] = *(const uint4*)(pa1 + 16);
      const uint4* pb = (const uint4*)(A.W2p + ((size_t)(kc * A.ntiles2 + w)) * 512 + lane * 16);
      bfr.u[0] = pb[0]; bfr.u[1] = pb[1];
      c0 = __builtin_amdgcn_wmma_f32_16x16x32_bf16(false, a0.v, false, bfr.v,
                                                   (short)0, c0, false, false);
      c1 = __builtin_amdgcn_wmma_f32_16x16x32_bf16(false, a1.v, false, bfr.v,
                                                   (short)0, c1, false, false);
    }
    const int col = w * 16 + lmod;
    if (col < A.Dout) {
      const float b = A.b2[col];
#pragma unroll
      for (int r = 0; r < 8; ++r) {
        int m = r + lhalf * 8;
        int rg0 = mbase + m;
        if (rg0 < A.M) {
          float v = c0[r] + b;
          if (A.res) v += A.res[(size_t)rg0 * A.ostride + col];
          A.out[(size_t)rg0 * A.ostride + col] = v;
        }
        int rg1 = mbase + m + 16;
        if (rg1 < A.M) {
          float v = c1[r] + b;
          if (A.res) v += A.res[(size_t)rg1 * A.ostride + col];
          A.out[(size_t)rg1 * A.ostride + col] = v;
        }
      }
    }
  }
}

// eb[i] = batch[row[i]]
__global__ __launch_bounds__(WG) void make_eb(const int* __restrict__ row,
                                              const int* __restrict__ batch,
                                              int* __restrict__ eb, int E) {
  int i = blockIdx.x * WG + threadIdx.x;
  if (i < E) eb[i] = batch[row[i]];
}

// agg[row[e], :] += e[e, :]   (32 lanes/edge, 4 floats/lane)
__global__ __launch_bounds__(WG) void scatter_add_k(const float* __restrict__ e,
                                                    const int* __restrict__ row,
                                                    float* __restrict__ agg, int E) {
  long long t = (long long)blockIdx.x * WG + threadIdx.x;
  int edge = (int)(t >> 5);
  if (edge >= E) return;
  int lane = (int)(t & 31);
  int r = row[edge];
  float4 v = *(const float4*)(e + (size_t)edge * 128 + lane * 4);
  float* dst = agg + (size_t)r * 128 + lane * 4;
  __hip_atomic_fetch_add(dst + 0, v.x, __ATOMIC_RELAXED, __HIP_MEMORY_SCOPE_AGENT);
  __hip_atomic_fetch_add(dst + 1, v.y, __ATOMIC_RELAXED, __HIP_MEMORY_SCOPE_AGENT);
  __hip_atomic_fetch_add(dst + 2, v.z, __ATOMIC_RELAXED, __HIP_MEMORY_SCOPE_AGENT);
  __hip_atomic_fetch_add(dst + 3, v.w, __ATOMIC_RELAXED, __HIP_MEMORY_SCOPE_AGENT);
}

// ---------------------------------------------------------------------------
// Host side
// ---------------------------------------------------------------------------
static inline size_t align256(size_t x) { return (x + 255) & ~(size_t)255; }

extern "C" void kernel_launch(void* const* d_in, const int* in_sizes, int n_in,
                              void* d_out, int out_size, void* d_ws, size_t ws_size,
                              hipStream_t stream) {
  (void)n_in; (void)out_size; (void)ws_size;
  const int Hh = 128;
  const int N = in_sizes[0] / 16;   // x: [N,16]
  const int E = in_sizes[1] / 4;    // edge_attr: [E,4]
  const int B = in_sizes[2] / 8;    // conditions: [B,8]

  const float* x    = (const float*)d_in[0];
  const float* eatt = (const float*)d_in[1];
  const float* cond = (const float*)d_in[2];

  // params flattened as sorted pytree:
  // cond_enc(3..6) decoder(7..10) edge_enc(11..14)
  // layers[i]: edge(15+8i..18+8i) node(19+8i..22+8i), each W1,W2,b1,b2
  // node_enc(47..50); then edge_index(51), batch(52)
  const float* ce_W1 = (const float*)d_in[3];
  const float* ce_W2 = (const float*)d_in[4];
  const float* ce_b1 = (const float*)d_in[5];
  const float* ce_b2 = (const float*)d_in[6];
  const float* de_W1 = (const float*)d_in[7];
  const float* de_W2 = (const float*)d_in[8];
  const float* de_b1 = (const float*)d_in[9];
  const float* de_b2 = (const float*)d_in[10];
  const float* ee_W1 = (const float*)d_in[11];
  const float* ee_W2 = (const float*)d_in[12];
  const float* ee_b1 = (const float*)d_in[13];
  const float* ee_b2 = (const float*)d_in[14];
  const float* ne_W1 = (const float*)d_in[47];
  const float* ne_W2 = (const float*)d_in[48];
  const float* ne_b1 = (const float*)d_in[49];
  const float* ne_b2 = (const float*)d_in[50];
  const int* edge_index = (const int*)d_in[51];
  const int* batch      = (const int*)d_in[52];
  const int* row = edge_index;
  const int* col = edge_index + E;

  // workspace carve
  char* wsb = (char*)d_ws;
  size_t off = 0;
  auto carve = [&](size_t bytes) -> char* {
    char* p = wsb + off; off = align256(off + bytes); return p;
  };
  float* h   = (float*)carve((size_t)N * Hh * 4);
  float* e   = (float*)carve((size_t)E * Hh * 4);
  float* agg = (float*)carve((size_t)N * Hh * 4);
  float* u   = (float*)carve((size_t)64 * Hh * 4);
  int*   eb  = (int*)carve((size_t)E * 4);
  unsigned short* pcur = (unsigned short*)carve((size_t)1000000 * 2);

  auto packW = [&](const float* W, int K, int Ncols) -> const unsigned short* {
    int Kpad = (K + 31) & ~31;
    int Ntiles = (Ncols + 15) >> 4;
    int nBlocks = (Kpad / 32) * Ntiles;
    unsigned short* dst = pcur;
    int grid = (nBlocks + (WG / 32) - 1) / (WG / 32);
    hipLaunchKernelGGL(pack_w_kernel, dim3(grid), dim3(WG), 0, stream,
                       W, dst, K, Ncols, Ntiles, nBlocks);
    pcur += (size_t)nBlocks * 512;
    return dst;
  };
  auto runMlp = [&](const MlpArgs& a) {
    int grid = (a.M + 31) / 32;
    hipLaunchKernelGGL(mlp2_kernel, dim3(grid), dim3(WG), 0, stream, a);
  };

  // pack all weights (tiny; L2-resident afterwards)
  const unsigned short* p_ne1 = packW(ne_W1, 16, 128);
  const unsigned short* p_ne2 = packW(ne_W2, 128, 128);
  const unsigned short* p_ee1 = packW(ee_W1, 4, 128);
  const unsigned short* p_ee2 = packW(ee_W2, 128, 128);
  const unsigned short* p_ce1 = packW(ce_W1, 8, 128);
  const unsigned short* p_ce2 = packW(ce_W2, 128, 128);
  const unsigned short* p_de1 = packW(de_W1, 128, 128);
  const unsigned short* p_de2 = packW(de_W2, 128, 6);
  const unsigned short *pLe1[4], *pLe2[4], *pLn1[4], *pLn2[4];
  const float *le_b1[4], *le_b2[4], *ln_b1[4], *ln_b2[4];
  for (int i = 0; i < 4; ++i) {
    pLe1[i]  = packW((const float*)d_in[15 + 8 * i], 512, 128);
    pLe2[i]  = packW((const float*)d_in[16 + 8 * i], 128, 128);
    le_b1[i] = (const float*)d_in[17 + 8 * i];
    le_b2[i] = (const float*)d_in[18 + 8 * i];
    pLn1[i]  = packW((const float*)d_in[19 + 8 * i], 384, 128);
    pLn2[i]  = packW((const float*)d_in[20 + 8 * i], 128, 128);
    ln_b1[i] = (const float*)d_in[21 + 8 * i];
    ln_b2[i] = (const float*)d_in[22 + 8 * i];
  }

  hipLaunchKernelGGL(make_eb, dim3((E + WG - 1) / WG), dim3(WG), 0, stream,
                     row, batch, eb, E);

  // encoders
  MlpArgs a{};
  a.seg[0] = {x, nullptr, 16, 32}; a.nseg = 1; a.M = N; a.Dpad = 32;
  a.W1p = p_ne1; a.b1 = ne_b1; a.W2p = p_ne2; a.b2 = ne_b2;
  a.Dout = 128; a.ntiles2 = 8; a.res = nullptr; a.out = h; a.ostride = 128;
  runMlp(a);

  a = MlpArgs{};
  a.seg[0] = {eatt, nullptr, 4, 32}; a.nseg = 1; a.M = E; a.Dpad = 32;
  a.W1p = p_ee1; a.b1 = ee_b1; a.W2p = p_ee2; a.b2 = ee_b2;
  a.Dout = 128; a.ntiles2 = 8; a.res = nullptr; a.out = e; a.ostride = 128;
  runMlp(a);

  a = MlpArgs{};
  a.seg[0] = {cond, nullptr, 8, 32}; a.nseg = 1; a.M = B; a.Dpad = 32;
  a.W1p = p_ce1; a.b1 = ce_b1; a.W2p = p_ce2; a.b2 = ce_b2;
  a.Dout = 128; a.ntiles2 = 8; a.res = nullptr; a.out = u; a.ostride = 128;
  runMlp(a);

  // message-passing layers
  for (int i = 0; i < 4; ++i) {
    // edge update: e = MLP([h[row], h[col], e, u[eb]])   (in-place on e)
    a = MlpArgs{};
    a.seg[0] = {h, row, 128, 128};
    a.seg[1] = {h, col, 128, 128};
    a.seg[2] = {e, nullptr, 128, 128};
    a.seg[3] = {u, eb, 128, 128};
    a.nseg = 4; a.M = E; a.Dpad = 512;
    a.W1p = pLe1[i]; a.b1 = le_b1[i]; a.W2p = pLe2[i]; a.b2 = le_b2[i];
    a.Dout = 128; a.ntiles2 = 8; a.res = nullptr; a.out = e; a.ostride = 128;
    runMlp(a);

    // aggregation: agg = scatter_add(e -> row)
    hipMemsetAsync(agg, 0, (size_t)N * Hh * 4, stream);
    hipLaunchKernelGGL(scatter_add_k,
                       dim3((int)(((long long)E * 32 + WG - 1) / WG)), dim3(WG),
                       0, stream, e, row, agg, E);

    // node update: h = h + MLP([h, agg, u[batch]])   (in-place on h)
    a = MlpArgs{};
    a.seg[0] = {h, nullptr, 128, 128};
    a.seg[1] = {agg, nullptr, 128, 128};
    a.seg[2] = {u, batch, 128, 128};
    a.nseg = 3; a.M = N; a.Dpad = 384;
    a.W1p = pLn1[i]; a.b1 = ln_b1[i]; a.W2p = pLn2[i]; a.b2 = ln_b2[i];
    a.Dout = 128; a.ntiles2 = 8; a.res = h; a.out = h; a.ostride = 128;
    runMlp(a);
  }

  // decoder: out = MLP(h)  [N, 6]
  a = MlpArgs{};
  a.seg[0] = {h, nullptr, 128, 128};
  a.nseg = 1; a.M = N; a.Dpad = 128;
  a.W1p = p_de1; a.b1 = de_b1; a.W2p = p_de2; a.b2 = de_b2;
  a.Dout = 6; a.ntiles2 = 1; a.res = nullptr; a.out = (float*)d_out; a.ostride = 6;
  runMlp(a);
}
